// RotaryMultiHeadedAttention_25658134627017
// MI455X (gfx1250) — matrix operand order repaired and checked by reading, not compiled, not run on hardware
//
#include <hip/hip_runtime.h>
#include <hip/hip_bf16.h>

// ---------------------------------------------------------------------------
// Rotary MHA for MI455X (gfx1250): bf16 WMMA (v_wmma_f32_16x16x32_bf16),
// fp32 accumulate, flash-attention inner loop with ASYNC global->LDS
// double-buffered K/V staging (ASYNCcnt), RoPE fused into Q/K GEMM epilogue.
// Wave32; fragment layouts per CDNA5 ISA 7.12.2.
// ---------------------------------------------------------------------------

typedef __attribute__((ext_vector_type(16))) __bf16 v16bf;
typedef __attribute__((ext_vector_type(8)))  __bf16 v8bf;
typedef __attribute__((ext_vector_type(8)))  float  v8f;
typedef __attribute__((ext_vector_type(4)))  int    v4i;

#define FP32_MIN_V (-3.4028234663852886e38f)
#define LN1E4_OVER_32 (0.28782313662425575f)   // ln(10000)/32

#if __has_builtin(__builtin_amdgcn_global_load_async_to_lds_b128) && \
    __has_builtin(__builtin_amdgcn_s_wait_asynccnt)
#define HAS_ASYNC_LDS 1
// prototype (from clang diagnostic): (v4i AS1*, v4i AS3*, Imm offset, Imm cpol)
#define ASYNC_CP16(gptr, lptr)                                              \
  __builtin_amdgcn_global_load_async_to_lds_b128(                           \
      (__attribute__((address_space(1))) v4i*)(gptr),                       \
      (__attribute__((address_space(3))) v4i*)(lptr), 0, 0)
#define ASYNC_WAIT0() __builtin_amdgcn_s_wait_asynccnt(0)
#else
#define HAS_ASYNC_LDS 0
#endif

__device__ __forceinline__ unsigned short f2bf(float f) {
  unsigned u = __float_as_uint(f);
  unsigned r = (u + 0x7FFFu + ((u >> 16) & 1u)) >> 16;   // RNE
  return (unsigned short)r;
}

__device__ __forceinline__ v16bf bcat(v8bf lo, v8bf hi) {
  return __builtin_shufflevector(lo, hi, 0,1,2,3,4,5,6,7,8,9,10,11,12,13,14,15);
}

__device__ __forceinline__ v8f wmma_bf16(v16bf a, v16bf b, v8f c) {
  return __builtin_amdgcn_wmma_f32_16x16x32_bf16(false, a, false, b,
                                                 (short)0, c, false, false);
}

// ---------------------------------------------------------------------------
// GEMM: Y[8192x512] = X[8192x512] * W^T + bias
// MODE 0: q-proj  -> RoPE -> *1/8 -> bf16 (B,H,T,64)
// MODE 1: k-proj  -> RoPE ->         bf16 (B,H,T,64)
// MODE 2: v-proj  ->                 bf16 (B,H,64,T)   (transposed)
// MODE 3: o-proj  (A is bf16 ctx) -> fp32 (B,T,512)
// block 256 = 8 waves; WG tile 128x64; wave tile 16x64 (4 C-tiles); BK=32.
// ---------------------------------------------------------------------------
template<int MODE>
__global__ __launch_bounds__(256)
void gemm_kernel(const float* __restrict__ Xf, const unsigned short* __restrict__ Xb,
                 const float* __restrict__ W,  const float* __restrict__ bias,
                 unsigned short* __restrict__ outB, float* __restrict__ outF)
{
  constexpr int F = 512;
  __shared__ unsigned short As[128][40];   // bf16, +8 pad
  __shared__ unsigned short Bs[64][40];

  const int tid  = threadIdx.x;
  const int wv   = tid >> 5;
  const int lane = tid & 31;
  const int m    = lane & 15;
  const int half = lane >> 4;
  const int m0   = blockIdx.x * 128;
  const int n0   = blockIdx.y * 64;

  v8f acc[4] = {};

  for (int k0 = 0; k0 < F; k0 += 32) {
    __syncthreads();
    if (MODE == 3) {                        // A already bf16
      const int lr = tid >> 2, lc = (tid & 3) * 8;
      #pragma unroll
      for (int it = 0; it < 2; ++it) {
        const int row = lr + 64 * it;
        uint4 v = *(const uint4*)&Xb[(size_t)(m0 + row) * F + k0 + lc];
        *(uint4*)&As[row][lc] = v;
      }
    } else {                                // A fp32 -> bf16
      const int lr = tid >> 3, lc = (tid & 7) * 4;
      #pragma unroll
      for (int it = 0; it < 4; ++it) {
        const int row = lr + 32 * it;
        float4 v = *(const float4*)&Xf[(size_t)(m0 + row) * F + k0 + lc];
        unsigned p0 = (unsigned)f2bf(v.x) | ((unsigned)f2bf(v.y) << 16);
        unsigned p1 = (unsigned)f2bf(v.z) | ((unsigned)f2bf(v.w) << 16);
        *(unsigned*)&As[row][lc]     = p0;
        *(unsigned*)&As[row][lc + 2] = p1;
      }
    }
    {                                       // W fp32 -> bf16
      const int lr = tid >> 3, lc = (tid & 7) * 4;
      #pragma unroll
      for (int it = 0; it < 2; ++it) {
        const int row = lr + 32 * it;
        float4 v = *(const float4*)&W[(size_t)(n0 + row) * F + k0 + lc];
        unsigned p0 = (unsigned)f2bf(v.x) | ((unsigned)f2bf(v.y) << 16);
        unsigned p1 = (unsigned)f2bf(v.z) | ((unsigned)f2bf(v.w) << 16);
        *(unsigned*)&Bs[row][lc]     = p0;
        *(unsigned*)&Bs[row][lc + 2] = p1;
      }
    }
    // prefetch next K-tile while this one is consumed
    if (k0 + 32 < F) {
      if (MODE == 3)
        __builtin_prefetch(&Xb[(size_t)(m0 + (tid >> 1)) * F + k0 + 32], 0, 3);
      else
        __builtin_prefetch(&Xf[(size_t)(m0 + (tid >> 1)) * F + k0 + 32], 0, 3);
      if (tid < 64)
        __builtin_prefetch(&W[(size_t)(n0 + tid) * F + k0 + 32], 0, 3);
    }
    __syncthreads();

    // A fragment (16x32): lane m holds K={8h..8h+7} and {16+8h..16+8h+7}
    v8bf alo = *(const v8bf*)&As[16 * wv + m][8 * half];
    v8bf ahi = *(const v8bf*)&As[16 * wv + m][16 + 8 * half];
    v16bf a  = bcat(alo, ahi);
    #pragma unroll
    for (int t = 0; t < 4; ++t) {           // B fragment: lane n holds K={8h..8h+7} and {16+8h..16+8h+7}
      v8bf blo = *(const v8bf*)&Bs[t * 16 + m][8 * half];
      v8bf bhi = *(const v8bf*)&Bs[t * 16 + m][16 + 8 * half];
      acc[t] = wmma_bf16(a, bcat(blo, bhi), acc[t]);
    }
  }

  // ----- epilogue: C element (M = r + 8*half, N = t*16 + (lane&15)) -----
  const int nl = lane & 15;
  if (MODE == 0 || MODE == 1) {
    const int bidx = m0 >> 10;              // batch (rope position = batch idx!)
    const int h    = blockIdx.y;            // head (BN==dk==64)
    const float scale = (MODE == 0) ? 0.125f : 1.0f;
    #pragma unroll
    for (int p = 0; p < 2; ++p) {           // tiles (p, p+2) pair cols j, j+32
      const int jlo = p * 16 + nl;
      const float th  = __expf(-(float)jlo * LN1E4_OVER_32);
      const float ang = (float)bidx * th;
      float sv, cv; __sincosf(ang, &sv, &cv);
      const float blo = bias[n0 + jlo];
      const float bhi = bias[n0 + jlo + 32];
      #pragma unroll
      for (int r = 0; r < 8; ++r) {
        float x = acc[p][r] + blo;
        float y = acc[p + 2][r] + bhi;
        float ro = (x * cv - y * sv) * scale;
        float rh = (y * cv + x * sv) * scale;
        const int tI = (m0 + 16 * wv + r + 8 * half) & 1023;
        size_t base = (((size_t)bidx * 8 + h) * 1024 + (size_t)tI) * 64;
        outB[base + jlo]      = f2bf(ro);
        outB[base + jlo + 32] = f2bf(rh);
      }
    }
  } else if (MODE == 2) {
    const int bidx = m0 >> 10;
    const int h    = blockIdx.y;
    #pragma unroll
    for (int t = 0; t < 4; ++t) {
      const int j = t * 16 + nl;
      const float bv = bias[n0 + j];
      #pragma unroll
      for (int r = 0; r < 8; ++r) {
        const int tI = (m0 + 16 * wv + r + 8 * half) & 1023;
        outB[(((size_t)bidx * 8 + h) * 64 + (size_t)j) * 1024 + tI] =
            f2bf(acc[t][r] + bv);
      }
    }
  } else {
    #pragma unroll
    for (int t = 0; t < 4; ++t) {
      const int j = n0 + t * 16 + nl;
      const float bv = bias[j];
      #pragma unroll
      for (int r = 0; r < 8; ++r) {
        const int grow = m0 + 16 * wv + r + 8 * half;
        outF[(size_t)grow * 512 + j] = acc[t][r] + bv;
      }
    }
  }
}

// ---------------------------------------------------------------------------
// Flash attention: grid (T/128, B*H). 8 waves x 16 query rows.
// Per 32-key block: S = Q*K^T (4 wmma), online softmax, O += P*V (4 wmma).
// K/V tiles staged with GLOBAL_LOAD_ASYNC_TO_LDS_B128 + ASYNCcnt, double
// buffered so DMA of block kb+1 overlaps compute of block kb.
// ---------------------------------------------------------------------------
__global__ __launch_bounds__(256)
void attn_kernel(const unsigned short* __restrict__ qg,
                 const unsigned short* __restrict__ kg,
                 const unsigned short* __restrict__ vg,
                 const int* __restrict__ maskp,
                 unsigned short* __restrict__ ctx)
{
#if HAS_ASYNC_LDS
  __shared__ unsigned short Qs[128][72];
  __shared__ unsigned short Ks[2][32][72];
  __shared__ unsigned short Vs[2][64][40];
#else
  __shared__ unsigned short Qs[128][72];
  __shared__ unsigned short Ks[1][32][72];
  __shared__ unsigned short Vs[1][64][40];
#endif
  __shared__ unsigned short Ps[8][16][40];   // per-wave P patch (C->A relayout)
  __shared__ int Msk[1024];

  const int tid  = threadIdx.x;
  const int wv   = tid >> 5;
  const int lane = tid & 31;
  const int m    = lane & 15;
  const int half = lane >> 4;
  const int bh   = blockIdx.y;
  const int b    = bh >> 3;
  const int h    = bh & 7;
  const int q0   = blockIdx.x * 128;

  const size_t qkbase = (size_t)bh * 1024 * 64;
  const size_t vbase  = (size_t)bh * 64 * 1024;

  const int lrK = tid >> 3, lcK = (tid & 7) * 8;   // 32-row x 64-col b128 tiling
  const int vrV = tid >> 2, vcV = (tid & 3) * 8;   // 64-row x 32-col b128 tiling

  // stage key mask row once (all key blocks reuse it from LDS)
  #pragma unroll
  for (int it = 0; it < 4; ++it)
    Msk[tid + 256 * it] = maskp[b * 1024 + tid + 256 * it];

#if HAS_ASYNC_LDS
  // async DMA: Q tile + first K/V block straight into LDS (no VGPR round trip)
  #pragma unroll
  for (int it = 0; it < 4; ++it)
    ASYNC_CP16(&qg[qkbase + (size_t)(q0 + lrK + 32 * it) * 64 + lcK],
               &Qs[lrK + 32 * it][lcK]);
  ASYNC_CP16(&kg[qkbase + (size_t)lrK * 64 + lcK], &Ks[0][lrK][lcK]);
  ASYNC_CP16(&vg[vbase + (size_t)vrV * 1024 + vcV], &Vs[0][vrV][vcV]);
  ASYNC_WAIT0();
  __syncthreads();
#else
  {
    #pragma unroll
    for (int it = 0; it < 4; ++it) {
      const int row = lrK + 32 * it;
      uint4 v = *(const uint4*)&qg[qkbase + (size_t)(q0 + row) * 64 + lcK];
      *(uint4*)&Qs[row][lcK] = v;
    }
  }
  __syncthreads();
#endif

  v16bf qa[2];
  #pragma unroll
  for (int ks = 0; ks < 2; ++ks) {
    v8bf lo = *(const v8bf*)&Qs[16 * wv + m][32 * ks + 8 * half];
    v8bf hi = *(const v8bf*)&Qs[16 * wv + m][32 * ks + 16 + 8 * half];
    qa[ks] = bcat(lo, hi);
  }

  v8f o[4] = {};
  float mrow[8], lrow[8];
  #pragma unroll
  for (int r = 0; r < 8; ++r) { mrow[r] = FP32_MIN_V; lrow[r] = 0.f; }

  for (int kb = 0; kb < 32; ++kb) {
#if HAS_ASYNC_LDS
    const int cur = kb & 1;
    if (kb + 1 < 32) {   // overlap next block's DMA with this block's compute
      ASYNC_CP16(&kg[qkbase + (size_t)((kb + 1) * 32 + lrK) * 64 + lcK],
                 &Ks[cur ^ 1][lrK][lcK]);
      ASYNC_CP16(&vg[vbase + (size_t)vrV * 1024 + (kb + 1) * 32 + vcV],
                 &Vs[cur ^ 1][vrV][vcV]);
    }
#else
    const int cur = 0;
    __syncthreads();
    {
      uint4 v = *(const uint4*)&kg[qkbase + (size_t)(kb * 32 + lrK) * 64 + lcK];
      *(uint4*)&Ks[0][lrK][lcK] = v;
      uint4 w = *(const uint4*)&vg[vbase + (size_t)vrV * 1024 + kb * 32 + vcV];
      *(uint4*)&Vs[0][vrV][vcV] = w;
    }
    __syncthreads();
#endif

    v8f s0 = {}, s1 = {};
    #pragma unroll
    for (int ks = 0; ks < 2; ++ks) {
      v8bf l0 = *(const v8bf*)&Ks[cur][m][32 * ks + 8 * half];
      v8bf h0 = *(const v8bf*)&Ks[cur][m][32 * ks + 16 + 8 * half];
      s0 = wmma_bf16(qa[ks], bcat(l0, h0), s0);
      v8bf l1 = *(const v8bf*)&Ks[cur][16 + m][32 * ks + 8 * half];
      v8bf h1 = *(const v8bf*)&Ks[cur][16 + m][32 * ks + 16 + 8 * half];
      s1 = wmma_bf16(qa[ks], bcat(l1, h1), s1);
    }

    const int mk0 = Msk[kb * 32 + m];
    const int mk1 = Msk[kb * 32 + 16 + m];

    #pragma unroll
    for (int r = 0; r < 8; ++r) {
      float a0 = mk0 ? s0[r] : FP32_MIN_V;
      float a1 = mk1 ? s1[r] : FP32_MIN_V;
      float mx = fmaxf(a0, a1);
      mx = fmaxf(mx, __shfl_xor(mx, 1, 16));
      mx = fmaxf(mx, __shfl_xor(mx, 2, 16));
      mx = fmaxf(mx, __shfl_xor(mx, 4, 16));
      mx = fmaxf(mx, __shfl_xor(mx, 8, 16));
      const float mn = fmaxf(mrow[r], mx);
      const float al = __expf(mrow[r] - mn);
      float p0 = mk0 ? __expf(a0 - mn) : 0.f;
      float p1 = mk1 ? __expf(a1 - mn) : 0.f;
      float rs = p0 + p1;
      rs += __shfl_xor(rs, 1, 16);
      rs += __shfl_xor(rs, 2, 16);
      rs += __shfl_xor(rs, 4, 16);
      rs += __shfl_xor(rs, 8, 16);
      lrow[r] = lrow[r] * al + rs;
      mrow[r] = mn;
      #pragma unroll
      for (int t = 0; t < 4; ++t) o[t][r] *= al;
      const int prow = r + 8 * half;
      Ps[wv][prow][m]      = f2bf(p0);       // C-layout -> LDS
      Ps[wv][prow][16 + m] = f2bf(p1);
    }

    // reload P in A-fragment layout (wave-local LDS; DS ops are in-order)
    v8bf plo = *(const v8bf*)&Ps[wv][m][8 * half];
    v8bf phi = *(const v8bf*)&Ps[wv][m][16 + 8 * half];
    v16bf pa = bcat(plo, phi);
    #pragma unroll
    for (int t = 0; t < 4; ++t) {
      v8bf blo = *(const v8bf*)&Vs[cur][t * 16 + m][8 * half];
      v8bf bhi = *(const v8bf*)&Vs[cur][t * 16 + m][16 + 8 * half];
      o[t] = wmma_bf16(pa, bcat(blo, bhi), o[t]);
    }

#if HAS_ASYNC_LDS
    if (kb + 1 < 32) { ASYNC_WAIT0(); __syncthreads(); }
#endif
  }

  #pragma unroll
  for (int r = 0; r < 8; ++r) {
    const float inv = lrow[r] > 0.f ? 1.f / lrow[r] : 0.f;
    const int tI = q0 + 16 * wv + r + 8 * half;
    #pragma unroll
    for (int t = 0; t < 4; ++t) {
      const int col = h * 64 + t * 16 + m;
      ctx[((size_t)b * 1024 + tI) * 512 + col] = f2bf(o[t][r] * inv);
    }
  }
}

// ---------------------------------------------------------------------------
extern "C" void kernel_launch(void* const* d_in, const int* in_sizes, int n_in,
                              void* d_out, int out_size, void* d_ws, size_t ws_size,
                              hipStream_t stream) {
  (void)in_sizes; (void)n_in; (void)out_size; (void)ws_size;
  const float* q    = (const float*)d_in[0];
  const float* k    = (const float*)d_in[1];
  const float* v    = (const float*)d_in[2];
  const int*   mask = (const int*)  d_in[3];
  const float* Wq   = (const float*)d_in[4];
  const float* bq   = (const float*)d_in[5];
  const float* Wk   = (const float*)d_in[6];
  const float* bk   = (const float*)d_in[7];
  const float* Wv   = (const float*)d_in[8];
  const float* bv   = (const float*)d_in[9];
  const float* Wo   = (const float*)d_in[10];
  const float* bo   = (const float*)d_in[11];

  unsigned short* ws = (unsigned short*)d_ws;
  const size_t NE = (size_t)8 * 8 * 1024 * 64;   // 4 Mi elements per buffer
  unsigned short* qb = ws;
  unsigned short* kb = ws + NE;
  unsigned short* vt = ws + 2 * NE;
  unsigned short* cx = ws + 3 * NE;

  dim3 grid(64, 8, 1), blk(256, 1, 1);
  gemm_kernel<0><<<grid, blk, 0, stream>>>(q, nullptr, Wq, bq, qb, nullptr);
  gemm_kernel<1><<<grid, blk, 0, stream>>>(k, nullptr, Wk, bk, kb, nullptr);
  gemm_kernel<2><<<grid, blk, 0, stream>>>(v, nullptr, Wv, bv, vt, nullptr);
  attn_kernel<<<dim3(8, 64, 1), blk, 0, stream>>>(qb, kb, vt, mask, cx);
  gemm_kernel<3><<<grid, blk, 0, stream>>>(nullptr, cx, Wo, bo, nullptr,
                                           (float*)d_out);
}
